// MultisourcesWindowedCrossAttention_32581621908174
// MI455X (gfx1250) — matrix operand order repaired and checked by reading, not compiled, not run on hardware
//
#include <hip/hip_runtime.h>
#include <hip/hip_bf16.h>

// ---------------- problem constants ----------------
#define B_    2
#define D_    256
#define NH_   8
#define N1_   256
#define N2_   576
#define NT_   832          // N1 + N2 windows
#define NTOK  64           // tokens per 8x8 window
#define EV_   32           // value channels per head
#define EPS_  1e-6f
#define SCALE_ 0.08838834764831845f   // 1/sqrt(128)

typedef __bf16          v16bf __attribute__((ext_vector_type(16)));
typedef unsigned short  v16us __attribute__((ext_vector_type(16)));
typedef float           v8f   __attribute__((ext_vector_type(8)));

// ---------------- helpers ----------------
static __device__ __forceinline__ unsigned short cvt_bf16(float f) {
  unsigned u = __builtin_bit_cast(unsigned, f);
  u += 0x7FFFu + ((u >> 16) & 1u);          // round-to-nearest-even
  return (unsigned short)(u >> 16);
}
static __device__ __forceinline__ float bf16f(unsigned short h) {
  return __builtin_bit_cast(float, ((unsigned)h) << 16);
}
static __device__ __forceinline__ v8f wmma_bf16(v16us a, v16us b, v8f c) {
  return __builtin_amdgcn_wmma_f32_16x16x32_bf16(
      false, __builtin_bit_cast(v16bf, a),
      false, __builtin_bit_cast(v16bf, b),
      (short)0, c, false, false);
}

// A-operand (16x32, 16-bit) lane layout used here (not the ISA table order):
// K = kb+half*16+0..15 (contiguous, matches B); per-lane row m = lane&15.
static __device__ __forceinline__ v16us loadA_f32(const float* row, int kbase, int half) {
  const float4* p0 = (const float4*)(row + kbase + half * 16);
  const float4* p1 = (const float4*)(row + kbase + 8 + half * 16);
  float4 f0 = p0[0], f1 = p0[1], f2 = p1[0], f3 = p1[1];
  v16us v;
  v[0]  = cvt_bf16(f0.x); v[1]  = cvt_bf16(f0.y); v[2]  = cvt_bf16(f0.z); v[3]  = cvt_bf16(f0.w);
  v[4]  = cvt_bf16(f1.x); v[5]  = cvt_bf16(f1.y); v[6]  = cvt_bf16(f1.z); v[7]  = cvt_bf16(f1.w);
  v[8]  = cvt_bf16(f2.x); v[9]  = cvt_bf16(f2.y); v[10] = cvt_bf16(f2.z); v[11] = cvt_bf16(f2.w);
  v[12] = cvt_bf16(f3.x); v[13] = cvt_bf16(f3.y); v[14] = cvt_bf16(f3.z); v[15] = cvt_bf16(f3.w);
  return v;
}
static __device__ __forceinline__ v16us loadA_bf16(const unsigned short* row, int kbase, int half) {
  const uint4* p0 = (const uint4*)(row + kbase + half * 16);
  const uint4* p1 = (const uint4*)(row + kbase + 8 + half * 16);
  uint4 a0 = p0[0], a1 = p1[0];
  unsigned t0[4] = {a0.x, a0.y, a0.z, a0.w};
  unsigned t1[4] = {a1.x, a1.y, a1.z, a1.w};
  v16us v;
#pragma unroll
  for (int i = 0; i < 4; i++) {
    v[2*i]     = (unsigned short)(t0[i] & 0xFFFFu);
    v[2*i + 1] = (unsigned short)(t0[i] >> 16);
    v[8 + 2*i] = (unsigned short)(t1[i] & 0xFFFFu);
    v[9 + 2*i] = (unsigned short)(t1[i] >> 16);
  }
  return v;
}
// B-operand (32x16, 16-bit): lane (l&15)=column; lanes 0-15 K=0..15, 16-31 K=16..31.
static __device__ __forceinline__ v16us loadB_w(const float* W, int kbase, int half, int col) {
  v16us v;
#pragma unroll
  for (int i = 0; i < 16; i++)
    v[i] = cvt_bf16(W[(size_t)(kbase + half * 16 + i) * 256 + col]);
  return v;
}

// window id -> pixel base pointer for token n
static __device__ __forceinline__ const float* pixel_ptr(const float* x1, const float* x2,
                                                         int b, int w, int n) {
  if (w < N1_) {
    int wh = w >> 4, ww = w & 15;
    int y = wh * 8 + (n >> 3), x = ww * 8 + (n & 7);
    return x1 + (((size_t)b * 128 + y) * 128 + x) * 256;
  } else {
    int w2 = w - N1_;
    int wh = w2 / 24, ww = w2 % 24;
    int y = wh * 8 + (n >> 3), x = ww * 8 + (n & 7);
    return x2 + (((size_t)b * 192 + y) * 192 + x) * 256;
  }
}

// ---------------- kernel 0: window mean ----------------
__global__ __launch_bounds__(256) void k_avg(const float* __restrict__ x1,
                                             const float* __restrict__ x2,
                                             float* __restrict__ avg) {
  int blk = blockIdx.x;               // b*NT + w
  int b = blk / NT_, w = blk % NT_;
  int d = threadIdx.x;
  float s = 0.f;
  for (int n = 0; n < NTOK; n++) s += pixel_ptr(x1, x2, b, w, n)[d];
  avg[(size_t)blk * 256 + d] = s * (1.f / 64.f);
}

// ---------------- kernel 1: qk = rmsnorm(avg@Wqk + b) -> q/k bf16 ----------------
__global__ __launch_bounds__(256) void k_qk(const float* __restrict__ avg,
                                            const float* __restrict__ Wqk,
                                            const float* __restrict__ bqk,
                                            const float* __restrict__ rmsw,
                                            unsigned short* __restrict__ qg,
                                            unsigned short* __restrict__ kg) {
  __shared__ float lds[16][256];
  __shared__ float sc[16];
  int wv = threadIdx.x >> 5, lane = threadIdx.x & 31;
  int half = lane >> 4, lm = lane & 15;
  int r0 = blockIdx.x * 16;
  const float* Arow = avg + (size_t)(r0 + lm) * 256;
  for (int tt = 0; tt < 2; tt++) {
    int c0 = wv * 16 + tt * 128;
    v8f acc = {};
#pragma unroll
    for (int kb = 0; kb < 8; kb++) {
      int kbase = kb * 32;
      v16us a = loadA_f32(Arow, kbase, half);
      v16us bm = loadB_w(Wqk, kbase, half, c0 + lm);
      acc = wmma_bf16(a, bm, acc);
    }
    float bias = bqk[c0 + lm];
#pragma unroll
    for (int j = 0; j < 8; j++) lds[j + half * 8][c0 + lm] = acc[j] + bias;
  }
  __syncthreads();
  if (threadIdx.x < 16) {
    int m = threadIdx.x;
    float s = 0.f;
    for (int c = 0; c < 256; c++) { float v = lds[m][c]; s += v * v; }
    sc[m] = rsqrtf(s * (1.f / 256.f) + EPS_);
  }
  __syncthreads();
  int ct = threadIdx.x;
  float rw = rmsw[ct];
  int h = ct >> 5, j = ct & 31;
  for (int m = 0; m < 16; m++) {
    int gr = r0 + m;
    int b = gr / NT_, w = gr % NT_;
    unsigned short bits = cvt_bf16(lds[m][ct] * sc[m] * rw);
    size_t base = ((size_t)(b * NH_ + h) * NT_ + w) * 16;
    if (j < 16) qg[base + j] = bits;
    else        kg[base + (j - 16)] = bits;
  }
}

// ---------------- kernel 2: V projection -> (b,h,win,tok,e) bf16 ----------------
// 16x32 tile per wave: one A fragment feeds two WMMAs.
__global__ __launch_bounds__(256) void k_vproj(const float* __restrict__ x1,
                                               const float* __restrict__ x2,
                                               const float* __restrict__ Wv,
                                               unsigned short* __restrict__ V) {
  int wv = threadIdx.x >> 5, lane = threadIdx.x & 31;
  int half = lane >> 4, lm = lane & 15;
  int t = blockIdx.x * 8 + wv;
  int rt = t >> 3, cp = t & 7;
  int r0 = rt * 16, c0 = cp * 32;
  int b   = r0 / (NT_ * NTOK);
  int rem = r0 % (NT_ * NTOK);
  int w = rem / NTOK, nb = rem % NTOK;
  const float* Arow = pixel_ptr(x1, x2, b, w, nb + lm);
  v8f acc0 = {}, acc1 = {};
#pragma unroll
  for (int kb = 0; kb < 8; kb++) {
    int kbase = kb * 32;
    v16us a  = loadA_f32(Arow, kbase, half);
    v16us b0 = loadB_w(Wv, kbase, half, c0 + lm);
    v16us b1 = loadB_w(Wv, kbase, half, c0 + 16 + lm);
    acc0 = wmma_bf16(a, b0, acc0);
    acc1 = wmma_bf16(a, b1, acc1);
  }
  int ca = c0 + lm,      ha = ca & 7, ea = ca >> 3;   // channel order (e H): c = e*8 + h
  int cb = c0 + 16 + lm, hb = cb & 7, eb = cb >> 3;
#pragma unroll
  for (int j = 0; j < 8; j++) {
    int nn = nb + j + half * 8;
    V[(((size_t)(b * NH_ + ha) * NT_ + w) * NTOK + nn) * EV_ + ea] = cvt_bf16(acc0[j]);
    V[(((size_t)(b * NH_ + hb) * NT_ + w) * NTOK + nn) * EV_ + eb] = cvt_bf16(acc1[j]);
  }
}

// ---------------- kernel 3: cross-source scores + softmax -> P bf16 ----------------
__global__ __launch_bounds__(256) void k_attn(const unsigned short* __restrict__ qg,
                                              const unsigned short* __restrict__ kg,
                                              unsigned short* __restrict__ P) {
  int wv = threadIdx.x >> 5, lane = threadIdx.x & 31;
  int idx = blockIdx.x * 8 + wv;              // (b*NH+h)*NT + q
  int bh = idx / NT_, q = idx % NT_;
  float qf[16];
  {
    const uint4* p = (const uint4*)(qg + ((size_t)bh * NT_ + q) * 16);
    uint4 a0 = p[0], a1 = p[1];
    unsigned tmp[8] = {a0.x, a0.y, a0.z, a0.w, a1.x, a1.y, a1.z, a1.w};
#pragma unroll
    for (int i = 0; i < 8; i++) {
      qf[2*i]     = bf16f((unsigned short)(tmp[i] & 0xFFFFu));
      qf[2*i + 1] = bf16f((unsigned short)(tmp[i] >> 16));
    }
  }
  int kw0 = (q < N1_) ? N1_ : 0;              // attend only to the other source
  int Nk  = (q < N1_) ? N2_ : N1_;
  int nr = Nk >> 5;                           // 18 or 8, exact multiples of 32
  float s[18];
  float mx = -1e30f;
  for (int r = 0; r < nr; r++) {
    int kk = r * 32 + lane;
    const uint4* p = (const uint4*)(kg + ((size_t)bh * NT_ + (kw0 + kk)) * 16);
    uint4 a0 = p[0], a1 = p[1];
    unsigned tmp[8] = {a0.x, a0.y, a0.z, a0.w, a1.x, a1.y, a1.z, a1.w};
    float dot = 0.f;
#pragma unroll
    for (int i = 0; i < 8; i++) {
      dot += qf[2*i]     * bf16f((unsigned short)(tmp[i] & 0xFFFFu));
      dot += qf[2*i + 1] * bf16f((unsigned short)(tmp[i] >> 16));
    }
    s[r] = dot * SCALE_;
    mx = fmaxf(mx, s[r]);
  }
#pragma unroll
  for (int off = 16; off > 0; off >>= 1) mx = fmaxf(mx, __shfl_xor(mx, off, 32));
  float sum = 0.f;
  for (int r = 0; r < nr; r++) { s[r] = __expf(s[r] - mx); sum += s[r]; }
#pragma unroll
  for (int off = 16; off > 0; off >>= 1) sum += __shfl_xor(sum, off, 32);
  float inv = 1.f / sum;
  for (int r = 0; r < nr; r++)
    P[((size_t)bh * NT_ + q) * N2_ + r * 32 + lane] = cvt_bf16(s[r] * inv);
}

// ---------------- kernel 4: O_h = P_h @ V_h -> pre-output (b,q,tok,c) bf16 ----------------
// 16x32 tile per wave: both e-halves of one token share the A fragment.
__global__ __launch_bounds__(256) void k_av(const unsigned short* __restrict__ P,
                                            const unsigned short* __restrict__ V,
                                            unsigned short* __restrict__ op) {
  int wv = threadIdx.x >> 5, lane = threadIdx.x & 31;
  int half = lane >> 4, lm = lane & 15;
  int t = blockIdx.x * 8 + wv;
  int bh = t / 3328;                           // 52 q-tiles * 64 token-columns
  int tt = t % 3328;
  int rt = tt / 64, n0 = tt % 64;              // token index = column block
  int q0 = rt * 16;
  int b = bh >> 3, h = bh & 7;
  int kw0 = (q0 < N1_) ? N1_ : 0;
  int Nk  = (q0 < N1_) ? N2_ : N1_;
  const unsigned short* Arow = P + ((size_t)bh * NT_ + q0 + lm) * N2_;
  const unsigned short* Vb0  = V + (((size_t)bh * NT_ + kw0) * NTOK + n0) * EV_ + lm;
  const unsigned short* Vb1  = Vb0 + 16;
  v8f acc0 = {}, acc1 = {};
  int nkb = Nk >> 5;
#pragma unroll 2
  for (int kb = 0; kb < nkb; kb++) {
    int kbase = kb * 32;
    if (kb + 2 < nkb) __builtin_prefetch(Arow + kbase + 64, 0, 3);
    v16us a = loadA_bf16(Arow, kbase, half);
    v16us b0, b1;
#pragma unroll
    for (int i = 0; i < 16; i++) {
      size_t koff = (size_t)(kbase + half * 16 + i) * (NTOK * EV_);
      b0[i] = Vb0[koff];
      b1[i] = Vb1[koff];
    }
    acc0 = wmma_bf16(a, b0, acc0);
    acc1 = wmma_bf16(a, b1, acc1);
  }
  int cOut0 = (lm) * 8 + h;                    // pre-W_vb channel order (e H)
  int cOut1 = (16 + lm) * 8 + h;
#pragma unroll
  for (int j = 0; j < 8; j++) {
    int q = q0 + j + half * 8;
    size_t base = (((size_t)b * NT_ + q) * NTOK + n0) * 256;
    op[base + cOut0] = cvt_bf16(acc0[j]);
    op[base + cOut1] = cvt_bf16(acc1[j]);
  }
}

// ---------------- kernel 5: out = op @ W_vb -> un-windowed f32 output ----------------
// 16x32 tile per wave.
__global__ __launch_bounds__(256) void k_final(const unsigned short* __restrict__ op,
                                               const float* __restrict__ Wvb,
                                               float* __restrict__ out) {
  int wv = threadIdx.x >> 5, lane = threadIdx.x & 31;
  int half = lane >> 4, lm = lane & 15;
  int t = blockIdx.x * 8 + wv;
  int rt = t >> 3, cp = t & 7;
  int r0 = rt * 16, c0 = cp * 32;
  int b   = r0 / (NT_ * NTOK);
  int rem = r0 % (NT_ * NTOK);
  int w = rem / NTOK, nb = rem % NTOK;
  const unsigned short* Arow = op + (size_t)(r0 + lm) * 256;
  v8f acc0 = {}, acc1 = {};
#pragma unroll
  for (int kb = 0; kb < 8; kb++) {
    int kbase = kb * 32;
    v16us a  = loadA_bf16(Arow, kbase, half);
    v16us b0 = loadB_w(Wvb, kbase, half, c0 + lm);
    v16us b1 = loadB_w(Wvb, kbase, half, c0 + 16 + lm);
    acc0 = wmma_bf16(a, b0, acc0);
    acc1 = wmma_bf16(a, b1, acc1);
  }
  int d0 = c0 + lm, d1 = c0 + 16 + lm;
#pragma unroll
  for (int j = 0; j < 8; j++) {
    int n = nb + j + half * 8;
    size_t off;
    if (w < N1_) {
      int wh = w >> 4, ww = w & 15;
      int y = wh * 8 + (n >> 3), x = ww * 8 + (n & 7);
      off = (((size_t)b * 128 + y) * 128 + x) * 256;
    } else {
      int w2 = w - N1_;
      int wh = w2 / 24, ww = w2 % 24;
      int y = wh * 8 + (n >> 3), x = ww * 8 + (n & 7);
      off = (size_t)(2 * 128 * 128 * 256) + (((size_t)b * 192 + y) * 192 + x) * 256;
    }
    out[off + d0] = acc0[j];
    out[off + d1] = acc1[j];
  }
}

// ---------------- launch ----------------
extern "C" void kernel_launch(void* const* d_in, const int* in_sizes, int n_in,
                              void* d_out, int out_size, void* d_ws, size_t ws_size,
                              hipStream_t stream) {
  (void)in_sizes; (void)n_in; (void)out_size; (void)ws_size;
  const float* x1   = (const float*)d_in[0];
  const float* x2   = (const float*)d_in[1];
  const float* Wqk  = (const float*)d_in[2];
  const float* bqk  = (const float*)d_in[3];
  const float* rmsw = (const float*)d_in[4];
  const float* Wv   = (const float*)d_in[5];
  const float* Wvb  = (const float*)d_in[6];
  float* out = (float*)d_out;

  // workspace layout (bytes), all 16B aligned
  char* ws = (char*)d_ws;
  float*          avg = (float*)(ws);                       //  1,703,936
  unsigned short* qg  = (unsigned short*)(ws +  1703936);   //    425,984
  unsigned short* kg  = (unsigned short*)(ws +  2129920);   //    425,984
  unsigned short* V   = (unsigned short*)(ws +  2555904);   // 54,525,952
  unsigned short* P   = (unsigned short*)(ws + 57081856);   // 15,335,424
  unsigned short* op  = (unsigned short*)(ws + 72417280);   // 54,525,952  (ends ~121 MB)

  k_avg  <<<B_ * NT_,            256, 0, stream>>>(x1, x2, avg);
  k_qk   <<<(B_ * NT_) / 16,     256, 0, stream>>>(avg, Wqk, bqk, rmsw, qg, kg);
  k_vproj<<<6656,                256, 0, stream>>>(x1, x2, Wv, V);
  k_attn <<<(B_ * NH_ * NT_)/8,  256, 0, stream>>>(qg, kg, P);
  k_av   <<<6656,                256, 0, stream>>>(P, V, op);
  k_final<<<6656,                256, 0, stream>>>(op, Wvb, out);
}
